// ACDMNET_71004399337964
// MI455X (gfx1250) — matrix operand order repaired and checked by reading, not compiled, not run on hardware
//
#include <hip/hip_runtime.h>
#include <hip/hip_bf16.h>

typedef __attribute__((ext_vector_type(16))) _Float16 v16h;
typedef __attribute__((ext_vector_type(8)))  float    v8f;

#define EMB   128
#define K_N   128
#define BATCH 4096

__device__ __forceinline__ float sigmoidf(float x) {
    return __builtin_amdgcn_rcpf(1.0f + __expf(-x));
}

// ---------------------------------------------------------------------------
// Phase 1: WMMA GEMMs.
//   job 0: S1[b,e]  = stu_v[stu_id[b]]  · |W1[e, 0:128]|            (4096x128)
//   job 1: E2[b,e]  = exer_v[exer_id[b]]· |W2[e, 0:128]|            (4096x128)
//   job 2: KB1[k,e] = knowledge_v[k]    · |W1[e, 128:256]| + b1[e]  ( 128x128)
//   job 3: KB2[k,e] = knowledge_v[k]    · |W2[e, 128:256]| + b2[e]  ( 128x128)
// One wave (block of 32) per 16-row stripe; 8 column tiles x 4 K-chunks.
// ---------------------------------------------------------------------------
__global__ __launch_bounds__(32)
void acdm_gemm_phase(const int* __restrict__ stu_id,
                     const int* __restrict__ exer_id,
                     const float* __restrict__ student_v,
                     const float* __restrict__ exercise_v,
                     const float* __restrict__ knowledge_v,
                     const float* __restrict__ W1,
                     const float* __restrict__ b1,
                     const float* __restrict__ W2,
                     const float* __restrict__ b2,
                     float* __restrict__ S1,
                     float* __restrict__ E2,
                     float* __restrict__ KB1,
                     float* __restrict__ KB2)
{
    int bt = blockIdx.x;
    const float* src; const int* ids; const float* W; const float* bias;
    float* out; int rowbase; int c0;
    if (bt < 256)      { src = student_v;   ids = stu_id;  W = W1; bias = nullptr; out = S1;  rowbase = bt*16;       c0 = 0;   }
    else if (bt < 512) { src = exercise_v;  ids = exer_id; W = W2; bias = nullptr; out = E2;  rowbase = (bt-256)*16; c0 = 0;   }
    else if (bt < 520) { src = knowledge_v; ids = nullptr; W = W1; bias = b1;      out = KB1; rowbase = (bt-512)*16; c0 = 128; }
    else               { src = knowledge_v; ids = nullptr; W = W2; bias = b2;      out = KB2; rowbase = (bt-520)*16; c0 = 128; }

    const int lane = threadIdx.x & 31;
    const int ml   = lane & 15;        // row within tile (A), column within tile (B/D)
    const int hi   = lane >> 4;        // lane half selects K sub-range / M+8

    const int  grow = rowbase + ml;
    const int  srow = ids ? ids[grow] : grow;
    const float* arow = src + (long)srow * EMB;

    v8f acc[8];
    const v8f vzero = {0.f,0.f,0.f,0.f,0.f,0.f,0.f,0.f};
    #pragma unroll
    for (int n = 0; n < 8; ++n) acc[n] = vzero;

    #pragma unroll
    for (int kk = 0; kk < 4; ++kk) {
        const int kbase = kk * 32;
        // A fragment (16x32 f16): lane half hi=0 covers K 0-15, hi=1 covers 16-31
        v16h a;
        #pragma unroll
        for (int i = 0; i < 8; ++i)  a[i]   = (_Float16)arow[kbase + hi*16 + i];
        #pragma unroll
        for (int i = 0; i < 8; ++i)  a[8+i] = (_Float16)arow[kbase + hi*16 + 8 + i];

        #pragma unroll
        for (int n = 0; n < 8; ++n) {
            const int ebase = n * 16;
            // B fragment (32x16 f16): element i holds K = hi*16 + i, N = ml
            // -> 16 consecutive floats of |W| per lane
            const float* wrow = W + (long)(ebase + ml) * 256 + c0 + kbase + hi*16;
            v16h bf;
            #pragma unroll
            for (int i = 0; i < 16; ++i) bf[i] = (_Float16)fabsf(wrow[i]);
            acc[n] = __builtin_amdgcn_wmma_f32_16x16x32_f16(
                         false, a, false, bf, (short)0, acc[n], false, false);
        }
    }

    // D layout: VGPR r -> M = r (+8 for lane half 1), N = ml
    #pragma unroll
    for (int n = 0; n < 8; ++n) {
        const int col = n*16 + ml;
        const float bv = bias ? bias[col] : 0.0f;
        #pragma unroll
        for (int r = 0; r < 8; ++r) {
            const int row = rowbase + (hi ? 8 + r : r);
            out[row * EMB + col] = acc[n][r] + bv;
        }
    }
}

// ---------------------------------------------------------------------------
// Phase 2: fused sigmoid / weighted reduce. Block handles 8 batch rows;
// KB tables live in LDS with 129-float row pitch (conflict-free: bank=(k+e)%64).
// Thread t -> (sub-batch = t>>7, k = t&127); e-reduction is register-local.
// ---------------------------------------------------------------------------
__global__ __launch_bounds__(256)
void acdm_main_phase(const int* __restrict__ stu_id,
                     const int* __restrict__ exer_id,
                     const float* __restrict__ kq,
                     const float* __restrict__ student_q,
                     const float* __restrict__ exercise_k,
                     const float* __restrict__ W3,
                     const float* __restrict__ b3,
                     const float* __restrict__ S1,
                     const float* __restrict__ E2,
                     const float* __restrict__ KB1,
                     const float* __restrict__ KB2,
                     float* __restrict__ out)
{
    extern __shared__ float sm[];
    float* KB1s = sm;                     // 128*129
    float* KB2s = KB1s + 128*129;         // 128*129
    float* s1s  = KB2s + 128*129;         // 256
    float* e2s  = s1s + 256;              // 256
    float* wv   = e2s + 256;              // 256
    float* redv = wv + 256;               // 8
    float* redc = redv + 8;               // 8

    const int tid = threadIdx.x;
    for (int idx = tid; idx < 128*128; idx += 256) {
        const int k = idx >> 7, e = idx & 127;
        KB1s[k*129 + e] = KB1[idx];
        KB2s[k*129 + e] = KB2[idx];
    }

    const int base = blockIdx.x * 8;
    const int sub  = tid >> 7;            // which of the 2 batch rows in this pass
    const int k    = tid & 127;
    const float b3v = b3[0];

    for (int g = 0; g < 4; ++g) {
        const int bpair = base + g*2;
        {   // stage per-b vectors: thread (sub,e=k) loads one element each
            const int b = bpair + sub;
            const int e = k;
            const int sid = stu_id[b], eid = exer_id[b];
            s1s[tid] = S1[b*EMB + e];
            e2s[tid] = E2[b*EMB + e];
            const float q = student_q[(long)sid*EMB + e] * exercise_k[(long)eid*EMB + e];
            wv[tid] = sigmoidf(q) * fabsf(W3[e]);
        }
        __syncthreads();

        const float* kb1r = KB1s + k*129;
        const float* kb2r = KB2s + k*129;
        const float* s1r  = s1s + sub*128;
        const float* e2r  = e2s + sub*128;
        const float* wvr  = wv  + sub*128;

        float acc = 0.0f;
        #pragma unroll 8
        for (int e = 0; e < 128; ++e) {
            const float t1 = sigmoidf(s1r[e] + kb1r[e]);
            const float t2 = sigmoidf(e2r[e] + kb2r[e]);
            acc += (t1 - t2) * wvr[e];
        }

        const int   b   = bpair + sub;
        float kqv = kq[b*K_N + k];
        float val = sigmoidf(acc + b3v) * kqv;

        // wave32 reduction over the 32 k's this wave owns
        #pragma unroll
        for (int off = 16; off > 0; off >>= 1) {
            val += __shfl_xor(val, off, 32);
            kqv += __shfl_xor(kqv, off, 32);
        }
        const int wave = tid >> 5;
        if ((tid & 31) == 0) { redv[wave] = val; redc[wave] = kqv; }
        __syncthreads();

        if (tid < 2) {   // waves 0-3 -> b (sub 0), waves 4-7 -> b (sub 1)
            float sv = 0.f, sc = 0.f;
            #pragma unroll
            for (int w = 0; w < 4; ++w) { sv += redv[tid*4 + w]; sc += redc[tid*4 + w]; }
            out[bpair + tid] = sv / sc;
        }
        __syncthreads();
    }
}

extern "C" void kernel_launch(void* const* d_in, const int* in_sizes, int n_in,
                              void* d_out, int out_size, void* d_ws, size_t ws_size,
                              hipStream_t stream) {
    const int*   stu_id      = (const int*)  d_in[0];
    const int*   exer_id     = (const int*)  d_in[1];
    const float* kq          = (const float*)d_in[2];
    const float* student_q   = (const float*)d_in[3];
    const float* exercise_k  = (const float*)d_in[4];
    const float* student_v   = (const float*)d_in[5];
    const float* exercise_v  = (const float*)d_in[6];
    const float* knowledge_v = (const float*)d_in[7];
    const float* W1          = (const float*)d_in[8];
    const float* b1          = (const float*)d_in[9];
    const float* W2          = (const float*)d_in[10];
    const float* b2          = (const float*)d_in[11];
    const float* W3          = (const float*)d_in[12];
    const float* b3          = (const float*)d_in[13];
    float* out = (float*)d_out;

    float* ws  = (float*)d_ws;
    float* S1  = ws;                          // 4096*128
    float* E2  = S1 + BATCH*EMB;              // 4096*128
    float* KB1 = E2 + BATCH*EMB;              // 128*128
    float* KB2 = KB1 + K_N*EMB;               // 128*128

    // Phase 1: all four GEMMs through WMMA. 528 row-tiles total.
    acdm_gemm_phase<<<528, 32, 0, stream>>>(
        stu_id, exer_id, student_v, exercise_v, knowledge_v,
        W1, b1, W2, b2, S1, E2, KB1, KB2);

    // Phase 2: fused sigmoid + weighted reduction.
    const int smemBytes = (2*128*129 + 3*256 + 16) * (int)sizeof(float); // ~135 KB
    hipFuncSetAttribute((const void*)acdm_main_phase,
                        hipFuncAttributeMaxDynamicSharedMemorySize, smemBytes);
    acdm_main_phase<<<BATCH/8, 256, smemBytes, stream>>>(
        stu_id, exer_id, kq, student_q, exercise_k, W3, b3,
        S1, E2, KB1, KB2, out);
}
